// GATMalwareClassifier_34471407517956
// MI455X (gfx1250) — matrix operand order repaired and checked by reading, not compiled, not run on hardware
//
#include <hip/hip_runtime.h>
#include <math.h>

#define H 4
#define C 32
#define HC 128
#define NEG_SLOPE 0.2f
#define BN_EPS 1e-5f
#define GATTR 4
#define LAYERS 3

typedef __attribute__((ext_vector_type(16))) __bf16 v16bf;
typedef __attribute__((ext_vector_type(8)))  __bf16 v8bf;
typedef __attribute__((ext_vector_type(8)))  float  v8f;

union FragBF { v16bf v; v8bf h2[2]; };

__device__ __forceinline__ unsigned fenc(float f) {
    unsigned u = __float_as_uint(f);
    return (u & 0x80000000u) ? ~u : (u | 0x80000000u);
}
__device__ __forceinline__ float fdec(unsigned k) {
    unsigned u = (k & 0x80000000u) ? (k & 0x7FFFFFFFu) : ~k;
    return __uint_as_float(u);
}

// ---------------- utility fills ----------------
__global__ void k_fill_u32(unsigned* __restrict__ p, unsigned v, size_t n) {
    size_t i = (size_t)blockIdx.x * blockDim.x + threadIdx.x;
    if (i < n) p[i] = v;
}

// ---------------- split f32 -> bf16 hi/lo ----------------
__global__ void k_split(const float* __restrict__ x, __bf16* __restrict__ hi,
                        __bf16* __restrict__ lo, size_t n) {
    size_t i = (size_t)blockIdx.x * blockDim.x + threadIdx.x;
    if (i >= n) return;
    float f = x[i];
    __bf16 h = (__bf16)f;
    hi[i] = h;
    lo[i] = (__bf16)(f - (float)h);
}

// ---------------- transpose+split weights: Wt[n][k] = W[k][n] ----------------
__global__ void k_wprep(const float* __restrict__ W, const float* __restrict__ R,
                        __bf16* __restrict__ wh, __bf16* __restrict__ wl,
                        __bf16* __restrict__ rh, __bf16* __restrict__ rl) {
    int i = blockIdx.x * blockDim.x + threadIdx.x;
    if (i >= HC * HC) return;
    int n = i >> 7, k = i & 127;
    float f = W[k * HC + n];
    __bf16 h = (__bf16)f;
    wh[i] = h; wl[i] = (__bf16)(f - (float)h);
    float g = R[k * HC + n];
    __bf16 gh = (__bf16)g;
    rh[i] = gh; rl[i] = (__bf16)(g - (float)gh);
}

// ---------------- dual WMMA GEMM: h = x@W, resid = x@res_W (K=128) ----------------
// block: 256 threads = 8 waves; wave w -> column tile [16w,16w+16); blockIdx.x -> 16-row tile
__global__ void k_gemm_dual(const __bf16* __restrict__ xhi, const __bf16* __restrict__ xlo,
                            const __bf16* __restrict__ wth, const __bf16* __restrict__ wtl,
                            const __bf16* __restrict__ rth, const __bf16* __restrict__ rtl,
                            float* __restrict__ hout, float* __restrict__ rout, int N) {
    int wave = threadIdx.x >> 5;
    int lane = threadIdx.x & 31;
    int half = lane >> 4;      // 0: lanes 0-15, 1: lanes 16-31
    int l16  = lane & 15;
    int row0 = blockIdx.x * 16;
    int col0 = wave * 16;

    int r = row0 + l16; if (r >= N) r = N - 1;   // clamp loads; stores predicated
    int cl = col0 + l16;

    const __bf16* xh = xhi + (size_t)r * HC;
    const __bf16* xl = xlo + (size_t)r * HC;
    const __bf16* bwh = wth + (size_t)cl * HC;
    const __bf16* bwl = wtl + (size_t)cl * HC;
    const __bf16* brh = rth + (size_t)cl * HC;
    const __bf16* brl = rtl + (size_t)cl * HC;

    v8f accW = {0.f,0.f,0.f,0.f,0.f,0.f,0.f,0.f};
    v8f accR = {0.f,0.f,0.f,0.f,0.f,0.f,0.f,0.f};

    for (int kb = 0; kb < HC; kb += 32) {
        // A fragment: lane holds 16 contiguous K starting at kb + half*16
        int kA0 = kb + half * 16;
        int kA1 = kb + half * 16 + 8;
        FragBF ah, al;
        ah.h2[0] = *(const v8bf*)(xh + kA0); ah.h2[1] = *(const v8bf*)(xh + kA1);
        al.h2[0] = *(const v8bf*)(xl + kA0); al.h2[1] = *(const v8bf*)(xl + kA1);
        // B fragment: lane holds 16 contiguous K starting at kb + half*16
        int kB = kb + half * 16;
        FragBF bh, bl, rh, rl;
        bh.h2[0] = *(const v8bf*)(bwh + kB); bh.h2[1] = *(const v8bf*)(bwh + kB + 8);
        bl.h2[0] = *(const v8bf*)(bwl + kB); bl.h2[1] = *(const v8bf*)(bwl + kB + 8);
        rh.h2[0] = *(const v8bf*)(brh + kB); rh.h2[1] = *(const v8bf*)(brh + kB + 8);
        rl.h2[0] = *(const v8bf*)(brl + kB); rl.h2[1] = *(const v8bf*)(brl + kB + 8);

        // near-fp32 via bf16 split: hi*hi + hi*lo + lo*hi
        accW = __builtin_amdgcn_wmma_f32_16x16x32_bf16(false, ah.v, false, bh.v, (short)0, accW, false, false);
        accW = __builtin_amdgcn_wmma_f32_16x16x32_bf16(false, ah.v, false, bl.v, (short)0, accW, false, false);
        accW = __builtin_amdgcn_wmma_f32_16x16x32_bf16(false, al.v, false, bh.v, (short)0, accW, false, false);
        accR = __builtin_amdgcn_wmma_f32_16x16x32_bf16(false, ah.v, false, rh.v, (short)0, accR, false, false);
        accR = __builtin_amdgcn_wmma_f32_16x16x32_bf16(false, ah.v, false, rl.v, (short)0, accR, false, false);
        accR = __builtin_amdgcn_wmma_f32_16x16x32_bf16(false, al.v, false, rh.v, (short)0, accR, false, false);
    }

    // C/D layout: VGPR i -> M = i (lanes 0-15) / 8+i (lanes 16-31), N = lane&15
    for (int i = 0; i < 8; ++i) {
        int row = row0 + half * 8 + i;
        int col = col0 + l16;
        if (row < N) {
            hout[(size_t)row * HC + col] = accW[i];
            rout[(size_t)row * HC + col] = accR[i];
        }
    }
}

// ---------------- per-node attention coefficients ----------------
__global__ void k_att(const float* __restrict__ h, const float* __restrict__ att_s,
                      const float* __restrict__ att_d, float* __restrict__ asrc,
                      float* __restrict__ adst, int N) {
    int i = blockIdx.x * blockDim.x + threadIdx.x;
    if (i >= N * H) return;
    int n = i >> 2, hd = i & 3;
    const float* hp = h + (size_t)n * HC + hd * C;
    const float* as = att_s + hd * C;
    const float* ad = att_d + hd * C;
    float s1 = 0.f, s2 = 0.f;
    for (int j = 0; j < C; ++j) { float v = hp[j]; s1 += v * as[j]; s2 += v * ad[j]; }
    asrc[i] = s1; adst[i] = s2;
}

// ---------------- edge pass 1: segment max (ordered-uint atomicMax) ----------------
__global__ void k_edge_max(const int* __restrict__ ei, const float* __restrict__ asrc,
                           const float* __restrict__ adst, unsigned* __restrict__ segmax,
                           int E, int N) {
    int i = blockIdx.x * blockDim.x + threadIdx.x;
    if (i >= (E + N) * H) return;
    int e = i >> 2, hd = i & 3;
    int s, d;
    if (e < E) { s = ei[e]; d = ei[E + e]; } else { s = d = e - E; }
    float v = asrc[s * H + hd] + adst[d * H + hd];
    v = v > 0.f ? v : NEG_SLOPE * v;
    atomicMax(&segmax[d * H + hd], fenc(v));
}

// ---------------- edge pass 2: exp + denom ----------------
__global__ void k_edge_exp(const int* __restrict__ ei, const float* __restrict__ asrc,
                           const float* __restrict__ adst, const unsigned* __restrict__ segmax,
                           float* __restrict__ exbuf, float* __restrict__ denom, int E, int N) {
    int i = blockIdx.x * blockDim.x + threadIdx.x;
    if (i >= (E + N) * H) return;
    int e = i >> 2, hd = i & 3;
    int s, d;
    if (e < E) { s = ei[e]; d = ei[E + e]; } else { s = d = e - E; }
    float v = asrc[s * H + hd] + adst[d * H + hd];
    v = v > 0.f ? v : NEG_SLOPE * v;
    float m = fdec(segmax[d * H + hd]);
    if (!isfinite(m)) m = 0.0f;
    float ex = expf(v - m);
    exbuf[i] = ex;
    atomicAdd(&denom[d * H + hd], ex);
}

// ---------------- edge pass 3: alpha-weighted scatter aggregate (wave per edge) ----------------
__global__ void k_edge_agg(const int* __restrict__ ei, const float* __restrict__ exbuf,
                           const float* __restrict__ denom, const float* __restrict__ h,
                           float* __restrict__ agg, int E, int N) {
    int wave = threadIdx.x >> 5, lane = threadIdx.x & 31;
    int e = blockIdx.x * 8 + wave;
    int Etot = E + N;
    if (e >= Etot) return;
    int s, d;
    if (e < E) { s = ei[e]; d = ei[E + e]; } else { s = d = e - E; }
    const float* hs = h + (size_t)s * HC;
    float* ag = agg + (size_t)d * HC;
    for (int p = 0; p < H; ++p) {
        float alpha = exbuf[e * H + p] / denom[d * H + p];
        int c = p * C + lane;
        atomicAdd(&ag[c], alpha * hs[c]);
    }
}

// ---------------- BN stats (LDS reduce, then global atomics) ----------------
__global__ void k_bnstats(const float* __restrict__ agg, const float* __restrict__ bias,
                          float* __restrict__ bnsum, float* __restrict__ bnsq, int N) {
    __shared__ float ls[HC], lq[HC];
    if (threadIdx.x < HC) { ls[threadIdx.x] = 0.f; lq[threadIdx.x] = 0.f; }
    __syncthreads();
    int total = N * HC;
    for (int i = blockIdx.x * blockDim.x + threadIdx.x; i < total; i += gridDim.x * blockDim.x) {
        int c = i & (HC - 1);
        float v = agg[i] + bias[c];
        atomicAdd(&ls[c], v);
        atomicAdd(&lq[c], v * v);
    }
    __syncthreads();
    if (threadIdx.x < HC) {
        atomicAdd(&bnsum[threadIdx.x], ls[threadIdx.x]);
        atomicAdd(&bnsq[threadIdx.x], lq[threadIdx.x]);
    }
}

__global__ void k_bnfin(const float* __restrict__ bnsum, const float* __restrict__ bnsq,
                        float* __restrict__ mu, float* __restrict__ rstd, int N) {
    int c = threadIdx.x;
    if (c >= HC) return;
    float m = bnsum[c] / (float)N;
    float v = bnsq[c] / (float)N - m * m;
    mu[c] = m;
    rstd[c] = rsqrtf(v + BN_EPS);
}

// ---------------- BN apply + ELU + residual + pooled scatter ----------------
__global__ void k_norm(const float* __restrict__ agg, const float* __restrict__ resid,
                       const float* __restrict__ bias, const float* __restrict__ gamma,
                       const float* __restrict__ beta, const float* __restrict__ mu,
                       const float* __restrict__ rstd, const int* __restrict__ batch,
                       float* __restrict__ xout, float* __restrict__ pooled, int N) {
    int i = blockIdx.x * blockDim.x + threadIdx.x;
    if (i >= N * HC) return;
    int n = i >> 7, c = i & 127;
    float v = agg[i] + bias[c];
    float b = (v - mu[c]) * rstd[c] * gamma[c] + beta[c];
    float a = b > 0.f ? b : expf(b) - 1.0f;
    float xn = a + resid[i];
    xout[i] = xn;
    atomicAdd(&pooled[(size_t)batch[n] * HC + c], xn);
}

__global__ void k_counts(const int* __restrict__ batch, float* __restrict__ counts, int N) {
    int i = blockIdx.x * blockDim.x + threadIdx.x;
    if (i < N) atomicAdd(&counts[batch[i]], 1.0f);
}

// ---------------- classifier ----------------
__global__ void k_clf_hidden(const float* __restrict__ pooled, const float* __restrict__ counts,
                             const float* __restrict__ gattr, const float* __restrict__ W1,
                             const float* __restrict__ b1, float* __restrict__ hdn, int B) {
    int i = blockIdx.x * blockDim.x + threadIdx.x;
    if (i >= B * HC) return;
    int b = i >> 7, j = i & 127;
    float inv = 1.0f / counts[b];
    float s = b1[j];
    for (int l = 0; l < LAYERS; ++l)
        for (int c = 0; c < HC; ++c)
            s += pooled[((size_t)l * B + b) * HC + c] * inv * W1[(l * HC + c) * HC + j];
    for (int g = 0; g < GATTR; ++g)
        s += gattr[b * GATTR + g] * W1[(LAYERS * HC + g) * HC + j];
    hdn[i] = s > 0.f ? s : expf(s) - 1.0f;
}

__global__ void k_clf_out(const float* __restrict__ hdn, const float* __restrict__ W2,
                          const float* __restrict__ b2, float* __restrict__ out, int B) {
    int i = blockIdx.x * blockDim.x + threadIdx.x;
    if (i >= B * 2) return;
    int b = i >> 1, o = i & 1;
    float s = b2[o];
    for (int j = 0; j < HC; ++j) s += hdn[b * HC + j] * W2[j * 2 + o];
    out[i] = s;
}

extern "C" void kernel_launch(void* const* d_in, const int* in_sizes, int n_in,
                              void* d_out, int out_size, void* d_ws, size_t ws_size,
                              hipStream_t stream) {
    const float* x0    = (const float*)d_in[0];
    const int*   ei    = (const int*)d_in[1];
    const int*   batch = (const int*)d_in[2];
    const float* gattr = (const float*)d_in[3];
    const int E = in_sizes[1] / 2;
    const int N = in_sizes[2];
    const int B = in_sizes[3] / GATTR;
    const float* clfW1 = (const float*)d_in[4 + 7 * LAYERS + 0];
    const float* clfb1 = (const float*)d_in[4 + 7 * LAYERS + 1];
    const float* clfW2 = (const float*)d_in[4 + 7 * LAYERS + 2];
    const float* clfb2 = (const float*)d_in[4 + 7 * LAYERS + 3];

    // workspace carve-out (256B aligned)
    char* ws = (char*)d_ws;
    size_t off = 0;
    auto alloc = [&](size_t bytes) -> void* {
        void* p = ws + off;
        off += (bytes + 255) & ~(size_t)255;
        return p;
    };
    const size_t NHC = (size_t)N * HC;
    const int Etot = E + N;

    float*  xbuf0  = (float*)alloc(NHC * 4);
    float*  xbuf1  = (float*)alloc(NHC * 4);
    __bf16* xhi    = (__bf16*)alloc(NHC * 2);
    __bf16* xlo    = (__bf16*)alloc(NHC * 2);
    float*  hbuf   = (float*)alloc(NHC * 4);
    float*  resid  = (float*)alloc(NHC * 4);
    float*  agg    = (float*)alloc(NHC * 4);
    float*  asrc   = (float*)alloc((size_t)N * H * 4);
    float*  adst   = (float*)alloc((size_t)N * H * 4);
    unsigned* segmax = (unsigned*)alloc((size_t)N * H * 4);
    float*  denom  = (float*)alloc((size_t)N * H * 4);
    float*  exbuf  = (float*)alloc((size_t)Etot * H * 4);
    __bf16* wth = (__bf16*)alloc(HC * HC * 2);
    __bf16* wtl = (__bf16*)alloc(HC * HC * 2);
    __bf16* rth = (__bf16*)alloc(HC * HC * 2);
    __bf16* rtl = (__bf16*)alloc(HC * HC * 2);
    float*  bnsum = (float*)alloc(2 * HC * 4);   // [sum | sumsq]
    float*  bnsq  = bnsum + HC;
    float*  mu    = (float*)alloc(HC * 4);
    float*  rstd  = (float*)alloc(HC * 4);
    float*  pooled = (float*)alloc((size_t)LAYERS * B * HC * 4);
    float*  counts = (float*)alloc((size_t)B * 4);
    float*  hdn    = (float*)alloc((size_t)B * HC * 4);

    auto cdiv = [](size_t a, size_t b) -> unsigned { return (unsigned)((a + b - 1) / b); };

    // once: zero pooled/counts, build counts
    k_fill_u32<<<cdiv((size_t)LAYERS * B * HC, 256), 256, 0, stream>>>((unsigned*)pooled, 0u, (size_t)LAYERS * B * HC);
    k_fill_u32<<<1, 256, 0, stream>>>((unsigned*)counts, 0u, (size_t)B);
    k_counts<<<cdiv(N, 256), 256, 0, stream>>>(batch, counts, N);

    const float* xin = x0;
    for (int l = 0; l < LAYERS; ++l) {
        const float* W     = (const float*)d_in[4 + 7 * l + 0];
        const float* att_s = (const float*)d_in[4 + 7 * l + 1];
        const float* att_d = (const float*)d_in[4 + 7 * l + 2];
        const float* bias  = (const float*)d_in[4 + 7 * l + 3];
        const float* gamma = (const float*)d_in[4 + 7 * l + 4];
        const float* beta  = (const float*)d_in[4 + 7 * l + 5];
        const float* resW  = (const float*)d_in[4 + 7 * l + 6];
        float* xout = (l % 2 == 0) ? xbuf0 : xbuf1;

        // per-layer scratch init
        k_fill_u32<<<cdiv(NHC, 256), 256, 0, stream>>>((unsigned*)agg, 0u, NHC);
        k_fill_u32<<<cdiv((size_t)N * H, 256), 256, 0, stream>>>((unsigned*)denom, 0u, (size_t)N * H);
        k_fill_u32<<<cdiv((size_t)N * H, 256), 256, 0, stream>>>(segmax, 0x007FFFFFu, (size_t)N * H); // enc(-inf)
        k_fill_u32<<<1, 256, 0, stream>>>((unsigned*)bnsum, 0u, (size_t)2 * HC);

        // GEMMs via WMMA
        k_split<<<cdiv(NHC, 256), 256, 0, stream>>>(xin, xhi, xlo, NHC);
        k_wprep<<<cdiv((size_t)HC * HC, 256), 256, 0, stream>>>(W, resW, wth, wtl, rth, rtl);
        k_gemm_dual<<<cdiv(N, 16), 256, 0, stream>>>(xhi, xlo, wth, wtl, rth, rtl, hbuf, resid, N);

        // attention + segment softmax + aggregate
        k_att<<<cdiv((size_t)N * H, 256), 256, 0, stream>>>(hbuf, att_s, att_d, asrc, adst, N);
        k_edge_max<<<cdiv((size_t)Etot * H, 256), 256, 0, stream>>>(ei, asrc, adst, segmax, E, N);
        k_edge_exp<<<cdiv((size_t)Etot * H, 256), 256, 0, stream>>>(ei, asrc, adst, segmax, exbuf, denom, E, N);
        k_edge_agg<<<cdiv(Etot, 8), 256, 0, stream>>>(ei, exbuf, denom, hbuf, agg, E, N);

        // BN + ELU + residual + pooling
        k_bnstats<<<2048, 256, 0, stream>>>(agg, bias, bnsum, bnsq, N);
        k_bnfin<<<1, HC, 0, stream>>>(bnsum, bnsq, mu, rstd, N);
        k_norm<<<cdiv(NHC, 256), 256, 0, stream>>>(agg, resid, bias, gamma, beta, mu, rstd,
                                                   batch, xout, pooled + (size_t)l * B * HC, N);
        xin = xout;
    }

    k_clf_hidden<<<cdiv((size_t)B * HC, 256), 256, 0, stream>>>(pooled, counts, gattr, clfW1, clfb1, hdn, B);
    k_clf_out<<<1, 128, 0, stream>>>(hdn, clfW2, clfb2, (float*)d_out, B);
}
